// Quant_Linear_21440476742421
// MI455X (gfx1250) — matrix operand order repaired and checked by reading, not compiled, not run on hardware
//
#include <hip/hip_runtime.h>

typedef __attribute__((ext_vector_type(16))) _Float16 v16h;
typedef __attribute__((ext_vector_type(8)))  _Float16 h8;
typedef __attribute__((ext_vector_type(4)))  _Float16 h4;
typedef __attribute__((ext_vector_type(8)))  float    v8f;
typedef __attribute__((ext_vector_type(4)))  float    f4;
typedef __attribute__((ext_vector_type(4)))  int      v4i;

#define TILE_M 128
#define TILE_N 128
#define BK     32
#define PITCH  40   // halfs per LDS row: 32 data + 8 pad (80B stride, bank-optimal b128)

// CDNA5 async global->LDS copy path (ASYNCcnt), with safe fallback.
#if defined(__has_builtin)
#if __has_builtin(__builtin_amdgcn_global_load_async_to_lds_b128)
#define USE_ASYNC_LDS 1
#endif
#endif

__device__ __forceinline__ void copy16_g2l(const _Float16* g, _Float16* l) {
#ifdef USE_ASYNC_LDS
  __builtin_amdgcn_global_load_async_to_lds_b128(
      (__attribute__((address_space(1))) v4i*)(g),
      (__attribute__((address_space(3))) v4i*)(l), 0, 0);
#else
  *reinterpret_cast<h8*>(l) = *reinterpret_cast<const h8*>(g);
#endif
}

__device__ __forceinline__ void wait_async_copies() {
#ifdef USE_ASYNC_LDS
#if __has_builtin(__builtin_amdgcn_s_wait_asynccnt)
  __builtin_amdgcn_s_wait_asynccnt(0);
#else
  asm volatile("s_wait_asynccnt 0x0" ::: "memory");
#endif
#endif
}

// ---------------------------------------------------------------------------
// Kernel 1: fp32 -> fp16 quantization (== float_quantize(x, exp=5, man=10):
// RNE to 10 mantissa bits, saturation at 65504, fp16 subnormal grid).
// NT loads: sources are single-use streams; keep L2 for the f16 matrices.
// ---------------------------------------------------------------------------
__global__ __launch_bounds__(256) void quantize_f32_to_f16(
    const float* __restrict__ src, _Float16* __restrict__ dst, int n4) {
  int i = blockIdx.x * blockDim.x + threadIdx.x;
  if (i >= n4) return;
  f4 v = __builtin_nontemporal_load(reinterpret_cast<const f4*>(src) + i);
  const float mx = 65504.0f;
  h4 r;
  r.x = (_Float16)fminf(fmaxf(v.x, -mx), mx);
  r.y = (_Float16)fminf(fmaxf(v.y, -mx), mx);
  r.z = (_Float16)fminf(fmaxf(v.z, -mx), mx);
  r.w = (_Float16)fminf(fmaxf(v.w, -mx), mx);
  reinterpret_cast<h4*>(dst)[i] = r;
}

// ---------------------------------------------------------------------------
// Kernel 2: C = quant_f16(A16 @ W16^T) + bias  via V_WMMA_F32_16X16X32_F16.
//   A16: M x K f16 row-major, W16: N x K f16 row-major (== B column-major),
//   C : M x N f32.
// 256 threads = 8 wave32, tile 128x128, BK=32, async double-buffered LDS.
// Wave grid 2(M) x 4(N); each wave computes a 4x2 grid of 16x16 WMMA tiles.
// ---------------------------------------------------------------------------
__global__ __launch_bounds__(256) void qgemm_f16_wmma(
    const _Float16* __restrict__ A,
    const _Float16* __restrict__ W,
    const float*    __restrict__ bias,
    float*          __restrict__ C,
    int M, int N, int K) {
  __shared__ _Float16 lsA[2][TILE_M * PITCH];
  __shared__ _Float16 lsB[2][TILE_N * PITCH];

  const int tid  = threadIdx.x;
  const int lane = tid & 31;
  const int wave = tid >> 5;
  const int wm   = wave >> 2;   // 0..1 -> 64 rows each
  const int wn   = wave & 3;    // 0..3 -> 32 cols each
  const int m16  = lane & 15;
  const int kh   = lane >> 4;   // which K-half this lane holds

  const int bm = blockIdx.y * TILE_M;
  const int bn = blockIdx.x * TILE_N;

  const _Float16* Ab = A + (size_t)bm * K;
  const _Float16* Wb = W + (size_t)bn * K;

  // Tile = 128 rows x 32 halfs = 512 x 16B chunks; each thread moves 2 chunks
  // per matrix per stage.
  const int c0 = tid;
  const int c1 = tid + 256;
  const int r0 = c0 >> 2, o0 = (c0 & 3) * 8;   // row, half-offset within row
  const int r1 = c1 >> 2, o1 = (c1 & 3) * 8;

  auto stage_tile = [&](int buf, int kk) {
    copy16_g2l(Ab + (size_t)r0 * K + kk + o0, &lsA[buf][r0 * PITCH + o0]);
    copy16_g2l(Ab + (size_t)r1 * K + kk + o1, &lsA[buf][r1 * PITCH + o1]);
    copy16_g2l(Wb + (size_t)r0 * K + kk + o0, &lsB[buf][r0 * PITCH + o0]);
    copy16_g2l(Wb + (size_t)r1 * K + kk + o1, &lsB[buf][r1 * PITCH + o1]);
  };

  // ---- stage 0 ----
  stage_tile(0, 0);
  wait_async_copies();
  __syncthreads();

  v8f acc[4][2] = {};

  const int ksteps = K / BK;
  for (int kt = 0; kt < ksteps; ++kt) {
    const int  cur  = kt & 1;
    const bool more = (kt + 1) < ksteps;

    // Fill the other buffer while this one is being consumed (async engine
    // overlaps with WMMA issue; previous iteration's barrier made it safe).
    if (more) stage_tile(cur ^ 1, (kt + 1) * BK);

    // ---- B fragments: lane holds K = kh*8+0..7 and 16+kh*8+0..7 of column n ----
    v16h bf[2];
#pragma unroll
    for (int tn = 0; tn < 2; ++tn) {
      const int row = wn * 32 + tn * 16 + m16;
      const _Float16* p = &lsB[cur][row * PITCH + kh * 8];
      h8 lo = *reinterpret_cast<const h8*>(p);
      h8 hi = *reinterpret_cast<const h8*>(p + 16);
      bf[tn] = __builtin_shufflevector(lo, hi, 0, 1, 2, 3, 4, 5, 6, 7,
                                               8, 9, 10, 11, 12, 13, 14, 15);
    }

    // ---- A fragments + WMMA: lane holds K = kh*8+0..7 and 16+kh*8+0..7 ----
#pragma unroll
    for (int tm = 0; tm < 4; ++tm) {
      const int row = wm * 64 + tm * 16 + m16;
      const _Float16* p = &lsA[cur][row * PITCH];
      h8 lo = *reinterpret_cast<const h8*>(p + kh * 8);
      h8 hi = *reinterpret_cast<const h8*>(p + 16 + kh * 8);
      v16h af = __builtin_shufflevector(lo, hi, 0, 1, 2, 3, 4, 5, 6, 7,
                                                8, 9, 10, 11, 12, 13, 14, 15);
#pragma unroll
      for (int tn = 0; tn < 2; ++tn) {
        acc[tm][tn] = __builtin_amdgcn_wmma_f32_16x16x32_f16(
            false, af, false, bf[tn], (short)0, acc[tm][tn], false, false);
      }
    }

    wait_async_copies();   // our portion of the next buffer has landed in LDS
    __syncthreads();       // everyone's portion has landed; reads of cur done
  }

  // ---- epilogue: quantize accumulator to fp16 grid, add bias, NT store ----
  // C/D layout: lanes 0-15 -> M = vgpr, N = lane; lanes 16-31 -> M = 8+vgpr.
#pragma unroll
  for (int tn = 0; tn < 2; ++tn) {
    const int n  = bn + wn * 32 + tn * 16 + m16;
    const float bv = bias[n];
#pragma unroll
    for (int tm = 0; tm < 4; ++tm) {
      const int mbase = bm + wm * 64 + tm * 16 + kh * 8;
#pragma unroll
      for (int r = 0; r < 8; ++r) {
        float v = acc[tm][tn][r];
        v = fminf(fmaxf(v, -65504.0f), 65504.0f);
        v = (float)(_Float16)v;                 // result quantization (e5m10)
        __builtin_nontemporal_store(v + bv, &C[(size_t)(mbase + r) * N + n]);
      }
    }
  }
}

// ---------------------------------------------------------------------------
extern "C" void kernel_launch(void* const* d_in, const int* in_sizes, int n_in,
                              void* d_out, int out_size, void* d_ws, size_t ws_size,
                              hipStream_t stream) {
  const float* x    = (const float*)d_in[0];   // M x K f32
  const float* w    = (const float*)d_in[1];   // N x K f32
  const float* bias = (const float*)d_in[2];   // N   f32
  // d_in[3] = exp (5), d_in[4] = man (10): e5m10 == IEEE fp16, hardwired.

  const int N = in_sizes[2];
  const int K = in_sizes[1] / N;
  const int M = in_sizes[0] / K;

  _Float16* a16 = (_Float16*)d_ws;             // 32 MB
  _Float16* w16 = a16 + (size_t)M * K;         // 32 MB

  const int n4a = (M * K) / 4;
  const int n4w = (N * K) / 4;
  quantize_f32_to_f16<<<(n4a + 255) / 256, 256, 0, stream>>>(x, a16, n4a);
  quantize_f32_to_f16<<<(n4w + 255) / 256, 256, 0, stream>>>(w, w16, n4w);

  dim3 grid(N / TILE_N, M / TILE_M);
  qgemm_f16_wmma<<<grid, 256, 0, stream>>>(a16, w16, bias, (float*)d_out, M, N, K);
}
